// GATBlock_20744692039827
// MI455X (gfx1250) — matrix operand order repaired and checked by reading, not compiled, not run on hardware
//
#include <hip/hip_runtime.h>
#include <hip/hip_bf16.h>

// ---------------------------------------------------------------------------
// GAT (3 layers, heads=1) + mean-pool + MLP head for gfx1250 (MI455X).
// GEMMs: v_wmma_f32_16x16x32_bf16 with bf16 hi/lo split (3 WMMAs / K-step,
// ~fp32 accuracy). Weights pre-packed (transposed bf16 hi/lo) once per call;
// A-strip staged in LDS once per block. Edge phase = scalar-score segment
// softmax + float4 atomic scatter (entire h/out working set is L2-resident).
// ---------------------------------------------------------------------------

typedef __attribute__((ext_vector_type(16))) __bf16 v16bf;
typedef __attribute__((ext_vector_type(8)))  __bf16 v8bf;
typedef __attribute__((ext_vector_type(8)))  float  v8f;

#define NEG_SLOPE 0.2f
#define KCHUNK 128
#define APAD 8   // LDS row padding (16B) to rotate banks

__device__ __forceinline__ void bf16_split(float a, __bf16& hi, __bf16& lo) {
  __bf16 h = (__bf16)a;
  hi = h;
  lo = (__bf16)(a - (float)h);
}

__device__ __forceinline__ v16bf cat8(v8bf a, v8bf b) {
  return __builtin_shufflevector(a, b, 0, 1, 2, 3, 4, 5, 6, 7,
                                 8, 9, 10, 11, 12, 13, 14, 15);
}

// One-shot weight pack: W[K x Ncols] f32 -> hi_t/lo_t[Ncols x K] bf16
// (transposed so a B-fragment lane reads its 16 K-values as two contiguous runs of 8).
__global__ void pack_weights(const float* __restrict__ W,
                             __bf16* __restrict__ hi_t,
                             __bf16* __restrict__ lo_t, int K, int Ncols) {
  const long long i = (long long)blockIdx.x * blockDim.x + threadIdx.x;
  if (i >= (long long)K * Ncols) return;
  const int k = (int)(i / Ncols);
  const int n = (int)(i % Ncols);
  __bf16 h, l;
  bf16_split(W[i], h, l);
  hi_t[(long long)n * K + k] = h;
  lo_t[(long long)n * K + k] = l;
}

// C = A(MxK) @ B(KxN) [+ bias] [relu]. A fp32 (row-major), B pre-packed
// bf16 hi/lo transposed [N x K]. block = (32,8): 8 waves, one 16x16 tile each;
// all waves share one M-strip staged in LDS as bf16 hi/lo.
__global__ void wmma_gemm_bf16x3(const float* __restrict__ A,
                                 const __bf16* __restrict__ Bhi,
                                 const __bf16* __restrict__ Blo,
                                 const float* __restrict__ bias,
                                 float* __restrict__ C,
                                 int M, int Ncols, int K, int relu) {
  __shared__ __bf16 shA[2][16][KCHUNK + APAD];  // [hi/lo][row][k]

  const int lane  = threadIdx.x;
  const int tid   = threadIdx.y * 32 + threadIdx.x;
  const int mtile = blockIdx.x;
  const int ntile = blockIdx.y * blockDim.y + threadIdx.y;
  const int row0  = mtile * 16;
  if (row0 >= M) return;                          // block-uniform exit
  const bool active = (ntile * 16 < Ncols);       // wave-uniform flag

  // per-lane fragment geometry (ISA 7.12.2, 16-bit A 16x32 / B 32x16)
  const int mrow  = lane & 15;
  const int abase = (lane < 16) ? 0 : 8;          // A k-offset for this half
  const int bbase = (lane < 16) ? 0 : 8;          // B k-offset for this half
  const long long bcol = (long long)(ntile * 16 + (lane & 15)) * K;

  v8f c = {};
  for (int kc = 0; kc < K; kc += KCHUNK) {
    const int klen = (K - kc < KCHUNK) ? (K - kc) : KCHUNK;
    __syncthreads();
    // cooperative fp32 load + split of the 16 x klen A-strip
    for (int idx = tid; idx < 16 * KCHUNK; idx += 256) {
      const int r = idx >> 7;        // / KCHUNK
      const int k = idx & (KCHUNK - 1);
      if (k < klen) {
        __bf16 h, l;
        bf16_split(A[(long long)(row0 + r) * K + kc + k], h, l);
        shA[0][r][k] = h;
        shA[1][r][k] = l;
      }
    }
    __syncthreads();

    if (active) {
      for (int k0 = 0; k0 < klen; k0 += 32) {
        // A fragment: two 16B LDS reads per hi/lo
        const __bf16* ah = &shA[0][mrow][k0 + abase];
        const __bf16* al = &shA[1][mrow][k0 + abase];
        const v16bf ahi = cat8(*(const v8bf*)ah, *(const v8bf*)(ah + 16));
        const v16bf alo = cat8(*(const v8bf*)al, *(const v8bf*)(al + 16));
        // B fragment: two 16B reads from packed weights (same k order as A)
        const __bf16* bh = Bhi + bcol + kc + k0 + bbase;
        const __bf16* bl = Blo + bcol + kc + k0 + bbase;
        const v16bf bhi = cat8(*(const v8bf*)bh, *(const v8bf*)(bh + 16));
        const v16bf blo = cat8(*(const v8bf*)bl, *(const v8bf*)(bl + 16));
        // c += alo*bhi + ahi*blo + ahi*bhi   (f32 accumulate)
        c = __builtin_amdgcn_wmma_f32_16x16x32_bf16(false, alo, false, bhi,
                                                    (short)0, c, false, false);
        c = __builtin_amdgcn_wmma_f32_16x16x32_bf16(false, ahi, false, blo,
                                                    (short)0, c, false, false);
        c = __builtin_amdgcn_wmma_f32_16x16x32_bf16(false, ahi, false, bhi,
                                                    (short)0, c, false, false);
      }
    }
  }

  if (active) {
    // C/D layout: VGPR j -> row = row0 + j + (lane>=16 ? 8 : 0), col = lane&15
    const int col   = ntile * 16 + (lane & 15);
    const int rbase = row0 + ((lane < 16) ? 0 : 8);
    const float bv  = bias ? bias[col] : 0.0f;
#pragma unroll
    for (int j = 0; j < 8; ++j) {
      float v = c[j] + bv;
      if (relu) v = fmaxf(v, 0.0f);
      C[(long long)(rbase + j) * Ncols + col] = v;
    }
  }
}

// Per-node attention scalars: ssrc[n] = h[n].a_src, sdst[n] = h[n].a_dst.
__global__ void node_scores(const float* __restrict__ h,
                            const float* __restrict__ asrc,
                            const float* __restrict__ adst,
                            float* __restrict__ ssrc, float* __restrict__ sdst,
                            int Nn, int D) {
  const int lane = threadIdx.x;
  const int node = blockIdx.x * blockDim.y + threadIdx.y;
  if (node >= Nn) return;
  const float* r = h + (long long)node * D;
  float s0 = 0.0f, s1 = 0.0f;
  for (int d = lane; d < D; d += 32) {
    const float v = r[d];
    s0 += v * asrc[d];
    s1 += v * adst[d];
  }
#pragma unroll
  for (int off = 16; off > 0; off >>= 1) {
    s0 += __shfl_xor(s0, off, 32);
    s1 += __shfl_xor(s1, off, 32);
  }
  if (lane == 0) { ssrc[node] = s0; sdst[node] = s1; }
}

__global__ void zero_u32(unsigned* __restrict__ p, long long n) {
  const long long i = (long long)blockIdx.x * blockDim.x + threadIdx.x;
  if (i < n) p[i] = 0u;
}

__device__ __forceinline__ void edge_fetch(const int* __restrict__ ei,
                                           long long i, int E, int& s, int& d) {
  if (i < E) { s = ei[i]; d = ei[(long long)E + i]; }
  else       { s = d = (int)(i - E); }               // self loop
}

__device__ __forceinline__ float edge_e(const float* ssrc, const float* sdst,
                                        int s, int d) {
  const float v = ssrc[s] + sdst[d];
  return v > 0.0f ? v : NEG_SLOPE * v;
}

// order-preserving float<->uint map for atomicMax
__device__ __forceinline__ unsigned f2key(float f) {
  const unsigned u = __float_as_uint(f);
  return (u & 0x80000000u) ? ~u : (u | 0x80000000u);
}
__device__ __forceinline__ float key2f(unsigned k) {
  const unsigned u = (k & 0x80000000u) ? (k ^ 0x80000000u) : ~k;
  return __uint_as_float(u);
}

__global__ void edge_max(const int* __restrict__ ei, int E, int Nn,
                         const float* __restrict__ ssrc,
                         const float* __restrict__ sdst,
                         unsigned* __restrict__ mkey) {
  const long long i = (long long)blockIdx.x * blockDim.x + threadIdx.x;
  if (i >= (long long)E + Nn) return;
  int s, d; edge_fetch(ei, i, E, s, d);
  atomicMax(&mkey[d], f2key(edge_e(ssrc, sdst, s, d)));
}

__global__ void edge_expsum(const int* __restrict__ ei, int E, int Nn,
                            const float* __restrict__ ssrc,
                            const float* __restrict__ sdst,
                            const unsigned* __restrict__ mkey,
                            float* __restrict__ denom) {
  const long long i = (long long)blockIdx.x * blockDim.x + threadIdx.x;
  if (i >= (long long)E + Nn) return;
  int s, d; edge_fetch(ei, i, E, s, d);
  atomicAdd(&denom[d], __expf(edge_e(ssrc, sdst, s, d) - key2f(mkey[d])));
}

// out[dst] += alpha * h[src]; one wave per edge, float4 per lane.
__global__ void edge_scatter(const int* __restrict__ ei, int E, int Nn,
                             const float* __restrict__ ssrc,
                             const float* __restrict__ sdst,
                             const unsigned* __restrict__ mkey,
                             const float* __restrict__ denom,
                             const float* __restrict__ h,
                             float* __restrict__ out, int D) {
  const int lane = threadIdx.x;
  const long long i = (long long)blockIdx.x * blockDim.y + threadIdx.y;
  if (i >= (long long)E + Nn) return;
  int s, d; edge_fetch(ei, i, E, s, d);
  const float alpha =
      __expf(edge_e(ssrc, sdst, s, d) - key2f(mkey[d])) / denom[d];
  const float4* hr = (const float4*)(h + (long long)s * D);
  float* od = out + (long long)d * D;
  for (int c = lane; c * 4 < D; c += 32) {
    const float4 v = hr[c];
    atomicAdd(&od[4 * c + 0], alpha * v.x);
    atomicAdd(&od[4 * c + 1], alpha * v.y);
    atomicAdd(&od[4 * c + 2], alpha * v.z);
    atomicAdd(&od[4 * c + 3], alpha * v.w);
  }
}

__global__ void add_bias_relu(float* __restrict__ out,
                              const float* __restrict__ b,
                              long long total, int D, int relu) {
  const long long i = (long long)blockIdx.x * blockDim.x + threadIdx.x;
  if (i >= total) return;
  float v = out[i] + b[(int)(i % D)];
  if (relu) v = fmaxf(v, 0.0f);
  out[i] = v;
}

__global__ void pool_accum(const float* __restrict__ h,
                           const int* __restrict__ batch,
                           float* __restrict__ pooled,
                           float* __restrict__ counts, int Nn, int D) {
  const int lane = threadIdx.x;
  const int node = blockIdx.x * blockDim.y + threadIdx.y;
  if (node >= Nn) return;
  const int b = batch[node];
  const float4* hr = (const float4*)(h + (long long)node * D);
  float* pr = pooled + (long long)b * D;
  for (int c = lane; c * 4 < D; c += 32) {
    const float4 v = hr[c];
    atomicAdd(&pr[4 * c + 0], v.x);
    atomicAdd(&pr[4 * c + 1], v.y);
    atomicAdd(&pr[4 * c + 2], v.z);
    atomicAdd(&pr[4 * c + 3], v.w);
  }
  if (lane == 0) atomicAdd(&counts[b], 1.0f);
}

// pooled = relu(pooled / max(counts,1))  (relu fused: pooled only used via relu)
__global__ void pool_finalize(float* __restrict__ pooled,
                              const float* __restrict__ counts,
                              int Bn, int D) {
  const int i = blockIdx.x * blockDim.x + threadIdx.x;
  if (i >= Bn * D) return;
  const float c = counts[i / D];
  pooled[i] = fmaxf(pooled[i] / fmaxf(c, 1.0f), 0.0f);
}

extern "C" void kernel_launch(void* const* d_in, const int* in_sizes, int n_in,
                              void* d_out, int out_size, void* d_ws,
                              size_t ws_size, hipStream_t stream) {
  (void)n_in; (void)ws_size;
  const float* x     = (const float*)d_in[0];
  const int*   ei    = (const int*)d_in[1];   // [2,E] flat: src then dst
  const int*   batch = (const int*)d_in[3];
  const float* asl[3] = {(const float*)d_in[5],  (const float*)d_in[9],  (const float*)d_in[13]};
  const float* adl[3] = {(const float*)d_in[6],  (const float*)d_in[10], (const float*)d_in[14]};
  const float* bl[3]  = {(const float*)d_in[7],  (const float*)d_in[11], (const float*)d_in[15]};
  const float* bf1 = (const float*)d_in[17];
  const float* bf2 = (const float*)d_in[19];

  const int Nn = in_sizes[3];      // batch has N elements
  const int E  = in_sizes[2];      // edge_weight has E elements
  const int D  = in_sizes[5];      // as1 has D elements
  const int H  = in_sizes[17];     // bf1 has H elements
  const int O  = in_sizes[19];     // bf2 has O elements
  const int Bn = out_size / O;

  // workspace carve-out (256B aligned regions)
  char* wp = (char*)d_ws;
  auto alloc = [&](size_t bytes) -> void* {
    void* r = (void*)wp;
    wp += (bytes + 255) & ~(size_t)255;
    return r;
  };
  float*    hbuf   = (float*)alloc((size_t)Nn * D * 4);  // h = x@W per layer
  float*    buf    = (float*)alloc((size_t)Nn * D * 4);  // layer output
  float*    ssrc   = (float*)alloc((size_t)Nn * 4);
  float*    sdst   = (float*)alloc((size_t)Nn * 4);
  unsigned* mkey   = (unsigned*)alloc((size_t)Nn * 4);
  float*    denom  = (float*)alloc((size_t)Nn * 4);
  float*    pooled = (float*)alloc((size_t)Bn * D * 4);
  float*    counts = (float*)alloc((size_t)Bn * 4);
  float*    hid    = (float*)alloc((size_t)Bn * H * 4);

  // packed weights: {W1,W2,W3,Wf1,Wf2} as transposed bf16 hi/lo
  const float* Wsrc[5] = {(const float*)d_in[4], (const float*)d_in[8],
                          (const float*)d_in[12], (const float*)d_in[16],
                          (const float*)d_in[18]};
  const int wK[5] = {D, D, D, D, H};
  const int wN[5] = {D, D, D, H, O};
  __bf16* Whi[5];
  __bf16* Wlo[5];
  for (int i = 0; i < 5; ++i) {
    Whi[i] = (__bf16*)alloc((size_t)wK[i] * wN[i] * 2);
    Wlo[i] = (__bf16*)alloc((size_t)wK[i] * wN[i] * 2);
  }

  const dim3 tb(32, 8, 1);                       // 8 wave32s per block
  const long long etot = (long long)E + Nn;      // + self loops
  const long long nd   = (long long)Nn * D;

  auto cdiv = [](long long a, long long b) -> unsigned {
    return (unsigned)((a + b - 1) / b);
  };

  for (int i = 0; i < 5; ++i) {
    const long long t = (long long)wK[i] * wN[i];
    pack_weights<<<dim3(cdiv(t, 256)), 256, 0, stream>>>(Wsrc[i], Whi[i],
                                                         Wlo[i], wK[i], wN[i]);
  }

  const float* in = x;
  for (int l = 0; l < 3; ++l) {
    // h = in @ W   (no bias, no relu; bias added post-scatter per reference)
    wmma_gemm_bf16x3<<<dim3(cdiv(Nn, 16), cdiv(D / 16, 8)), tb, 0, stream>>>(
        in, Whi[l], Wlo[l], nullptr, hbuf, Nn, D, D, 0);
    node_scores<<<dim3(cdiv(Nn, 8)), tb, 0, stream>>>(hbuf, asl[l], adl[l],
                                                      ssrc, sdst, Nn, D);
    zero_u32<<<dim3(cdiv(nd, 256)), 256, 0, stream>>>((unsigned*)buf, nd);
    zero_u32<<<dim3(cdiv(Nn, 256)), 256, 0, stream>>>(mkey, Nn);
    zero_u32<<<dim3(cdiv(Nn, 256)), 256, 0, stream>>>((unsigned*)denom, Nn);
    edge_max<<<dim3(cdiv(etot, 256)), 256, 0, stream>>>(ei, E, Nn, ssrc, sdst,
                                                        mkey);
    edge_expsum<<<dim3(cdiv(etot, 256)), 256, 0, stream>>>(ei, E, Nn, ssrc,
                                                           sdst, mkey, denom);
    edge_scatter<<<dim3(cdiv(etot, 8)), tb, 0, stream>>>(
        ei, E, Nn, ssrc, sdst, mkey, denom, hbuf, buf, D);
    add_bias_relu<<<dim3(cdiv(nd, 256)), 256, 0, stream>>>(buf, bl[l], nd, D,
                                                           l > 0 ? 1 : 0);
    in = buf;
  }

  // global mean pool + fused relu
  zero_u32<<<dim3(cdiv((long long)Bn * D, 256)), 256, 0, stream>>>(
      (unsigned*)pooled, (long long)Bn * D);
  zero_u32<<<dim3(cdiv(Bn, 256)), 256, 0, stream>>>((unsigned*)counts, Bn);
  pool_accum<<<dim3(cdiv(Nn, 8)), tb, 0, stream>>>(buf, batch, pooled, counts,
                                                   Nn, D);
  pool_finalize<<<dim3(cdiv((long long)Bn * D, 256)), 256, 0, stream>>>(
      pooled, counts, Bn, D);

  // MLP head: hid = relu(pooled @ Wf1 + bf1); out = hid @ Wf2 + bf2
  wmma_gemm_bf16x3<<<dim3(cdiv(Bn, 16), cdiv(H / 16, 8)), tb, 0, stream>>>(
      pooled, Whi[3], Wlo[3], bf1, hid, Bn, H, D, 1);
  wmma_gemm_bf16x3<<<dim3(cdiv(Bn, 16), cdiv(O / 16, 8)), tb, 0, stream>>>(
      hid, Whi[4], Wlo[4], bf2, (float*)d_out, Bn, O, H, 0);
}
